// MakeMoE_66073776881834
// MI455X (gfx1250) — matrix operand order repaired and checked by reading, not compiled, not run on hardware
//
#include <hip/hip_runtime.h>

typedef __attribute__((ext_vector_type(8)))  float  v8f;
typedef __attribute__((ext_vector_type(8)))  __bf16 v8bf;
typedef __attribute__((ext_vector_type(16))) __bf16 v16bf;

#define Bq 8192
#define Dq 2048
#define Eq 8
#define TM 64                 // rows per block tile (4 x 16 WMMA subtiles)
#define TN 128                // cols per block tile (8 waves x 16 cols)
#define KC 64                 // K chunk staged in LDS (2 x K32 WMMA steps)
#define LDA 72                // LDS A row stride in bf16 (bank stride 36 -> conflict-free)
#define PERM_CAP (Bq + Eq * TM)   // 8704 worst-case padded rows
#define MT (PERM_CAP / TM)        // 136 static M-tiles (graph-capture safe)
#define NKB (Dq / 32)             // 64 K-blocks of 32

// ---------- prep kernels: group tokens by expert, pad segments to TM ----------

__global__ void moe_init(int* counts, int* fill, int* perm) {
  int i = blockIdx.x * blockDim.x + threadIdx.x;
  if (i < PERM_CAP) perm[i] = -1;
  if (i < Eq) { counts[i] = 0; fill[i] = 0; }
}

__global__ void moe_count(const int* __restrict__ idx, int* counts) {
  int i = blockIdx.x * blockDim.x + threadIdx.x;
  if (i < Bq) atomicAdd(&counts[idx[i] & (Eq - 1)], 1);
}

__global__ void moe_scan(const int* __restrict__ counts, int* offs) {
  if (threadIdx.x == 0) {
    int acc = 0;
    offs[0] = 0;
    for (int e = 0; e < Eq; ++e) {
      acc += ((counts[e] + TM - 1) / TM) * TM;
      offs[e + 1] = acc;
    }
  }
}

__global__ void moe_scatter(const int* __restrict__ idx, const int* __restrict__ offs,
                            int* fill, int* perm) {
  int i = blockIdx.x * blockDim.x + threadIdx.x;
  if (i < Bq) {
    int e = idx[i] & (Eq - 1);
    int p = atomicAdd(&fill[e], 1);
    perm[offs[e] + p] = i;
  }
}

__device__ __forceinline__ void split_bf16(float f, __bf16& h, __bf16& l) {
  h = (__bf16)f;                       // RNE, top ~8 mantissa bits
  l = (__bf16)(f - (float)h);          // next ~8 bits (residual exact in f32)
}

// ---- W pre-split: one 64B record per (e, kblock, col) = 32 K-contiguous bf16 ----
// record element addr = ((e*NKB + kb)*Dq + n)*32 + k'   (k' = 0..31)

__global__ __launch_bounds__(256)
void moe_wsplit(const float* __restrict__ W, __bf16* __restrict__ Whi,
                __bf16* __restrict__ Wlo) {
  __shared__ __bf16 Lh[128 * 40];      // stride 40 bf16 = 80B (16B-multiple)
  __shared__ __bf16 Ll[128 * 40];
  const int t  = threadIdx.x;
  const int n0 = (blockIdx.x & 15) * 128;
  const int kb = (blockIdx.x >> 4) & (NKB - 1);
  const int e  = blockIdx.x >> 10;
  const float* src = W + ((size_t)e * Dq + kb * 32) * Dq + n0;
  #pragma unroll
  for (int s = 0; s < 16; ++s) {
    int idx = s * 256 + t;             // 0..4095 over 32 rows x 128 cols
    int row = idx >> 7;
    int col = idx & 127;
    float f = src[(size_t)row * Dq + col];   // coalesced 512B row segments
    __bf16 h, l; split_bf16(f, h, l);
    Lh[col * 40 + row] = h;
    Ll[col * 40 + row] = l;
  }
  __syncthreads();
  const size_t rbase = ((size_t)(e * NKB + kb) * Dq + n0) * 32;
  if (t < 128) {                       // dump hi records, coalesced 64B/thread
    const v8bf* s8 = (const v8bf*)&Lh[t * 40];
    v8bf* d8 = (v8bf*)(Whi + rbase + (size_t)t * 32);
    d8[0] = s8[0]; d8[1] = s8[1]; d8[2] = s8[2]; d8[3] = s8[3];
  } else {                             // dump lo records
    int c = t - 128;
    const v8bf* s8 = (const v8bf*)&Ll[c * 40];
    v8bf* d8 = (v8bf*)(Wlo + rbase + (size_t)c * 32);
    d8[0] = s8[0]; d8[1] = s8[1]; d8[2] = s8[2]; d8[3] = s8[3];
  }
}

// ------- grouped GEMM via 3-term bf16 split: out[rows] = x[rows] @ W[e] + b[e] -------

template <bool PRESPLIT>
__global__ __launch_bounds__(256)
void moe_gemm(const float* __restrict__ x, const float* __restrict__ W,
              const __bf16* __restrict__ Whi, const __bf16* __restrict__ Wlo,
              const float* __restrict__ bias, const int* __restrict__ offs,
              const int* __restrict__ perm, float* __restrict__ out) {
  __shared__ __bf16 Ah[TM * LDA];
  __shared__ __bf16 Al[TM * LDA];
  __shared__ int    tids[TM];
  __shared__ int    s_exp;

  const int r0 = blockIdx.y * TM;
  const int total = offs[Eq];
  if (r0 >= total) return;                     // uniform whole-block exit

  const int t = threadIdx.x;
  if (t < TM) tids[t] = perm[r0 + t];
  if (t == 0) {
    int e = 0;
    while (r0 >= offs[e + 1]) ++e;
    s_exp = e;
  }
  __syncthreads();
  const int e = s_exp;

  const int lane = t & 31;
  const int wave = t >> 5;
  const int hi   = lane >> 4;
  const int lo   = lane & 15;
  const int n    = blockIdx.x * TN + wave * 16 + lo;

  const float*  wp = W + (size_t)e * Dq * Dq + n;                       // fallback path
  const size_t  rb0 = ((size_t)e * NKB * Dq + n) * 32 + 16 * hi;        // presplit path
  v8f acc[4] = {v8f{}, v8f{}, v8f{}, v8f{}};

  const int arow = t >> 2;
  const int aseg = (t & 3) * 16;

  for (int k0 = 0; k0 < Dq; k0 += KC) {
    // ---- stage A tile as bf16 hi/lo (gathered rows; padding rows -> zeros) ----
    {
      int id = tids[arow];
      const float4* src = (const float4*)(x + (size_t)(id < 0 ? 0 : id) * Dq + k0 + aseg);
      float4 z = make_float4(0.f, 0.f, 0.f, 0.f);
      __bf16 hb[16], lb[16];
      #pragma unroll
      for (int j = 0; j < 4; ++j) {
        float4 v = (id >= 0) ? src[j] : z;
        split_bf16(v.x, hb[4*j+0], lb[4*j+0]);
        split_bf16(v.y, hb[4*j+1], lb[4*j+1]);
        split_bf16(v.z, hb[4*j+2], lb[4*j+2]);
        split_bf16(v.w, hb[4*j+3], lb[4*j+3]);
      }
      v8bf* dh = (v8bf*)&Ah[arow * LDA + aseg];
      v8bf* dl = (v8bf*)&Al[arow * LDA + aseg];
      dh[0] = *(v8bf*)&hb[0]; dh[1] = *(v8bf*)&hb[8];
      dl[0] = *(v8bf*)&lb[0]; dl[1] = *(v8bf*)&lb[8];
    }
    __syncthreads();

    #pragma unroll
    for (int kt = 0; kt < KC / 32; ++kt) {     // one K=32 WMMA step
      const int kb = (k0 >> 5) + kt;
      v16bf bh, bl;
      if constexpr (PRESPLIT) {
        // fragment-native records: two 16B loads each, 32B-aligned, wave spans 1KB
        const __bf16* rh = Whi + rb0 + (size_t)kb * Dq * 32;
        const __bf16* rl = Wlo + rb0 + (size_t)kb * Dq * 32;
        v8bf h0 = *(const v8bf*)rh, h1 = *(const v8bf*)(rh + 8);
        v8bf l0 = *(const v8bf*)rl, l1 = *(const v8bf*)(rl + 8);
        bh = __builtin_shufflevector(h0, h1, 0,1,2,3,4,5,6,7,8,9,10,11,12,13,14,15);
        bl = __builtin_shufflevector(l0, l1, 0,1,2,3,4,5,6,7,8,9,10,11,12,13,14,15);
      } else {
        const float* wq = wp + (size_t)(kb * 32 + 16 * hi) * Dq;
        #pragma unroll
        for (int j = 0; j < 16; ++j) {
          __bf16 h, l;
          split_bf16(wq[(size_t)j * Dq], h, l);
          bh[j] = h; bl[j] = l;
        }
      }

      #pragma unroll
      for (int m = 0; m < 4; ++m) {
        const int abase = (m * 16 + lo) * LDA + kt * 32 + 16 * hi;
        v8bf ah0 = *(const v8bf*)&Ah[abase];
        v8bf ah1 = *(const v8bf*)&Ah[abase + 8];
        v8bf al0 = *(const v8bf*)&Al[abase];
        v8bf al1 = *(const v8bf*)&Al[abase + 8];
        v16bf ah = __builtin_shufflevector(ah0, ah1, 0,1,2,3,4,5,6,7,8,9,10,11,12,13,14,15);
        v16bf al = __builtin_shufflevector(al0, al1, 0,1,2,3,4,5,6,7,8,9,10,11,12,13,14,15);

        acc[m] = __builtin_amdgcn_wmma_f32_16x16x32_bf16(
            false, al, false, bh, (short)0, acc[m], false, false);
        acc[m] = __builtin_amdgcn_wmma_f32_16x16x32_bf16(
            false, ah, false, bl, (short)0, acc[m], false, false);
        acc[m] = __builtin_amdgcn_wmma_f32_16x16x32_bf16(
            false, ah, false, bh, (short)0, acc[m], false, false);
      }
    }
    __syncthreads();
  }

  // ---- bias + scattered store (C row = v + 8*hi per ISA layout) ----
  const float bv = bias[(size_t)e * Dq + n];
  #pragma unroll
  for (int m = 0; m < 4; ++m) {
    #pragma unroll
    for (int v = 0; v < 8; ++v) {
      int rid = tids[m * 16 + v + 8 * hi];
      if (rid >= 0) out[(size_t)rid * Dq + n] = acc[m][v] + bv;
    }
  }
}

// ---------------------------------- launch ----------------------------------

extern "C" void kernel_launch(void* const* d_in, const int* in_sizes, int n_in,
                              void* d_out, int out_size, void* d_ws, size_t ws_size,
                              hipStream_t stream) {
  const float* x    = (const float*)d_in[0];
  const float* W    = (const float*)d_in[1];
  const float* bias = (const float*)d_in[2];
  const int*   idx  = (const int*)d_in[3];
  float* out = (float*)d_out;

  char* ws = (char*)d_ws;
  int* counts = (int*)ws;                 // 8
  int* offs   = counts + 8;               // 9
  int* fill   = counts + 17;              // 8
  int* perm   = counts + 32;              // PERM_CAP

  const size_t int_bytes = (size_t)(32 + PERM_CAP) * sizeof(int);
  const size_t walign    = (int_bytes + 255) & ~(size_t)255;
  const size_t welems    = (size_t)Eq * Dq * Dq;            // 33.5M per buffer
  const bool   presplit  = ws_size >= walign + 2 * welems * sizeof(__bf16);
  __bf16* Whi = (__bf16*)(ws + walign);
  __bf16* Wlo = Whi + welems;

  moe_init   <<<(PERM_CAP + 255) / 256, 256, 0, stream>>>(counts, fill, perm);
  moe_count  <<<(Bq + 255) / 256,       256, 0, stream>>>(idx, counts);
  moe_scan   <<<1,                       32, 0, stream>>>(counts, offs);
  moe_scatter<<<(Bq + 255) / 256,       256, 0, stream>>>(idx, offs, fill, perm);

  dim3 grid(Dq / TN, MT);                 // 16 x 136, static
  if (presplit) {
    moe_wsplit<<<Eq * NKB * (Dq / 128), 256, 0, stream>>>(W, Whi, Wlo);
    moe_gemm<true><<<grid, 256, 0, stream>>>(x, W, Whi, Wlo, bias, offs, perm, out);
  } else {
    moe_gemm<false><<<grid, 256, 0, stream>>>(x, W, Whi, Wlo, bias, offs, perm, out);
  }
}
